// ConvBlock_80607946211552
// MI455X (gfx1250) — matrix operand order repaired and checked by reading, not compiled, not run on hardware
//
#include <hip/hip_runtime.h>

// ---------------------------------------------------------------------------
// GraphSAGE 2-layer + JumpingKnowledge(cat) + linear, for MI455X (gfx1250).
// f32 aggregation / atomics (L2-resident), f16 WMMA GEMMs with f32 accum.
// Weights staged in LDS (bank-skewed) once per block: B-fragments feed the
// WMMAs via strided 2-byte LDS reads instead of per-wave global reloads.
// ---------------------------------------------------------------------------

typedef __attribute__((ext_vector_type(16))) _Float16 v16h;
typedef __attribute__((ext_vector_type(8)))  float    v8f;
typedef __attribute__((ext_vector_type(4)))  float    v4f;

#define C128 128
#define LW_STRIDE 136            // halves per LDS row (272B): 16B-aligned, +4 banks/row skew
#define LW_MAT    (C128 * LW_STRIDE)

// ---------------- utility kernels ----------------

__global__ void k_zero(float* __restrict__ p, long long n) {
  long long i = (long long)blockIdx.x * blockDim.x + threadIdx.x;
  if (i < n) p[i] = 0.0f;
}

// transpose + convert weights: WT[k*OUT + n] = (f16) W[n*K + k]
__global__ void k_prep_wt(const float* __restrict__ W, _Float16* __restrict__ WT,
                          int K, int OUT) {
  int idx = blockIdx.x * blockDim.x + threadIdx.x;
  if (idx >= K * OUT) return;
  int k = idx / OUT, n = idx - k * OUT;
  WT[k * OUT + n] = (_Float16)W[n * K + k];
}

// in-degree (float so it feeds the divide directly)
__global__ void k_count(const int* __restrict__ dst, float* __restrict__ cnt, int E) {
  int e = blockIdx.x * blockDim.x + threadIdx.x;
  if (e < E) atomicAdd(&cnt[dst[e]], 1.0f);
}

// scatter-add: one thread per (edge, 4-channel chunk)
__global__ void k_scatter(const float* __restrict__ src_feat,
                          const int* __restrict__ src, const int* __restrict__ dst,
                          float* __restrict__ agg, int E) {
  long long tid = (long long)blockIdx.x * blockDim.x + threadIdx.x;
  int e  = (int)(tid >> 5);
  int c4 = (int)((tid & 31) << 2);
  if (e >= E) return;
  int s = src[e], d = dst[e];
  v4f v = *(const v4f*)(src_feat + (long long)s * C128 + c4);
  float* ap = agg + (long long)d * C128 + c4;
  atomicAdd(ap + 0, v.x);
  atomicAdd(ap + 1, v.y);
  atomicAdd(ap + 2, v.z);
  atomicAdd(ap + 3, v.w);
}

// agg[n][c] /= max(cnt[n], 1)
__global__ void k_normalize(float* __restrict__ agg, const float* __restrict__ cnt,
                            long long total) {
  long long i = (long long)blockIdx.x * blockDim.x + threadIdx.x;
  if (i >= total) return;
  int n = (int)(i >> 7);
  float c = cnt[n];
  agg[i] *= 1.0f / fmaxf(c, 1.0f);
}

// ---------------- WMMA GEMM ----------------
// A-fragment (16-bit A 16x32 ISA layout): lane m = L%16 is the matrix row,
// hi = L/16 selects K-halves; halves 0..7 = K kb+hi*8.., halves 8..15 = K kb+16+hi*8..
__device__ __forceinline__ v16h load_a_frag(const float* __restrict__ A,
                                            int row, int kb, int hi) {
  const float* p = A + (long long)row * C128 + kb + hi * 8;
  v4f f0 = *(const v4f*)(p);
  v4f f1 = *(const v4f*)(p + 4);
  v4f f2 = *(const v4f*)(p + 16);
  v4f f3 = *(const v4f*)(p + 20);
  v16h a;
#pragma unroll
  for (int i = 0; i < 4; ++i) {
    a[i]      = (_Float16)f0[i];
    a[4 + i]  = (_Float16)f1[i];
    a[8 + i]  = (_Float16)f2[i];
    a[12 + i] = (_Float16)f3[i];
  }
  return a;
}

// out[16-node tile][128] = relu(A1 @ W1T + A2 @ W2T + bias)
// W*T are f16, row-major [K][128 N] (pre-transposed). Both 128x128 weight
// matrices are staged into LDS once per block; a B-fragment (
// 16 halves = K rows, lane = N col) is 16 strided 2-byte LDS reads.
__global__ void __launch_bounds__(256)
k_gemm_dual(const float* __restrict__ A1, const _Float16* __restrict__ W1T,
            const float* __restrict__ A2, const _Float16* __restrict__ W2T,
            const float* __restrict__ bias, float* __restrict__ out, int nTiles) {
  __shared__ _Float16 lw[2 * LW_MAT];

  // ---- stage weights into LDS (all 256 threads, 16B vectors) ----
#pragma unroll
  for (int i = 0; i < 8; ++i) {
    int idx = threadIdx.x + i * 256;          // 2048 uint4 per matrix
    int row = idx >> 4;
    int vec = idx & 15;
    *(uint4*)(&lw[row * LW_STRIDE + vec * 8]) =
        *(const uint4*)(W1T + row * C128 + vec * 8);
  }
#pragma unroll
  for (int i = 0; i < 8; ++i) {
    int idx = threadIdx.x + i * 256;
    int row = idx >> 4;
    int vec = idx & 15;
    *(uint4*)(&lw[LW_MAT + row * LW_STRIDE + vec * 8]) =
        *(const uint4*)(W2T + row * C128 + vec * 8);
  }
  __syncthreads();

  int lane = threadIdx.x & 31;
  int wave = threadIdx.x >> 5;
  int tile = blockIdx.x * 8 + wave;
  if (tile < nTiles) {          // wave-uniform: EXEC all-ones inside
    int nbase = tile * 16;
    int m  = lane & 15;
    int hi = lane >> 4;
    int row = nbase + m;

    // 8 independent accumulators (one per 16-wide output column tile)
    v8f acc[8];
#pragma unroll
    for (int nt = 0; nt < 8; ++nt) {
      float bv = bias[nt * 16 + m];
      acc[nt] = (v8f){bv, bv, bv, bv, bv, bv, bv, bv};
    }

#pragma unroll
    for (int kc = 0; kc < 4; ++kc) {
      const int kb = kc * 32;
      // A-fragments for this K-step only (global, streaming)
      v16h a1 = load_a_frag(A1, row, kb, hi);
      v16h a2 = load_a_frag(A2, row, kb, hi);

      const _Float16* lrow1 = lw + (kb + hi * 8) * LW_STRIDE + m;
      const _Float16* lrow2 = lrow1 + LW_MAT;

      // ---- W1 pass: B-fragments from LDS, 8 independent WMMAs
#pragma unroll
      for (int nt = 0; nt < 8; ++nt) {
        v16h b;
#pragma unroll
        for (int s = 0; s < 16; ++s) b[s] = lrow1[(s + (s >> 3) * 8) * LW_STRIDE + nt * 16];
        acc[nt] = __builtin_amdgcn_wmma_f32_16x16x32_f16(false, a1, false, b,
                                                         (short)0, acc[nt], false, false);
      }
      // ---- W2 pass
#pragma unroll
      for (int nt = 0; nt < 8; ++nt) {
        v16h b;
#pragma unroll
        for (int s = 0; s < 16; ++s) b[s] = lrow2[(s + (s >> 3) * 8) * LW_STRIDE + nt * 16];
        acc[nt] = __builtin_amdgcn_wmma_f32_16x16x32_f16(false, a2, false, b,
                                                         (short)0, acc[nt], false, false);
      }
    }

    // relu + store (lanes 0..15 cover row nbase+r, lanes 16..31 row nbase+r+8:
    // stores coalesce into 64B segments)
#pragma unroll
    for (int nt = 0; nt < 8; ++nt) {
#pragma unroll
      for (int r = 0; r < 8; ++r) {
        float v = acc[nt][r];
        v = v > 0.0f ? v : 0.0f;
        out[(long long)(nbase + r + 8 * hi) * C128 + nt * 16 + m] = v;
      }
    }
  }
}

// ---------------- host launcher ----------------

extern "C" void kernel_launch(void* const* d_in, const int* in_sizes, int n_in,
                              void* d_out, int out_size, void* d_ws, size_t ws_size,
                              hipStream_t stream) {
  const float* x     = (const float*)d_in[0];
  const int*   ei    = (const int*)d_in[1];
  const float* Wl1   = (const float*)d_in[2];
  const float* Wr1   = (const float*)d_in[3];
  const float* b1    = (const float*)d_in[4];
  const float* Wl2   = (const float*)d_in[5];
  const float* Wr2   = (const float*)d_in[6];
  const float* b2    = (const float*)d_in[7];
  const float* Wlin  = (const float*)d_in[8];
  const float* blin  = (const float*)d_in[9];
  float* out = (float*)d_out;

  const int N = in_sizes[0] / C128;
  const int E = in_sizes[1] / 2;
  const int* src = ei;
  const int* dst = ei + E;

  // workspace carve-out (256B aligned slices)
  size_t off = 0;
  char* base = (char*)d_ws;
  auto take = [&](size_t bytes) -> void* {
    void* p = base + off;
    off += (bytes + 255) & ~(size_t)255;
    return p;
  };
  const long long NF = (long long)N * C128;
  float* agg = (float*)take(NF * 4);
  float* h1  = (float*)take(NF * 4);
  float* h2  = (float*)take(NF * 4);
  float* cnt = (float*)take((size_t)N * 4);
  _Float16* WT1l  = (_Float16*)take(C128 * C128 * 2);
  _Float16* WT1r  = (_Float16*)take(C128 * C128 * 2);
  _Float16* WT2l  = (_Float16*)take(C128 * C128 * 2);
  _Float16* WT2r  = (_Float16*)take(C128 * C128 * 2);
  _Float16* WTlin = (_Float16*)take(256 * C128 * 2);  // [256 K][128 N]

  const int B = 256;
  dim3 blk(B);

  // 1) weight prep (transpose + f16)
  k_prep_wt<<<dim3((C128 * C128 + B - 1) / B), blk, 0, stream>>>(Wl1, WT1l, C128, C128);
  k_prep_wt<<<dim3((C128 * C128 + B - 1) / B), blk, 0, stream>>>(Wr1, WT1r, C128, C128);
  k_prep_wt<<<dim3((C128 * C128 + B - 1) / B), blk, 0, stream>>>(Wl2, WT2l, C128, C128);
  k_prep_wt<<<dim3((C128 * C128 + B - 1) / B), blk, 0, stream>>>(Wr2, WT2r, C128, C128);
  k_prep_wt<<<dim3((256 * C128 + B - 1) / B), blk, 0, stream>>>(Wlin, WTlin, 256, C128);

  // 2) degree counts (shared by both layers)
  k_zero<<<dim3((N + B - 1) / B), blk, 0, stream>>>(cnt, N);
  k_count<<<dim3((E + B - 1) / B), blk, 0, stream>>>(dst, cnt, E);

  const long long scatThreads = (long long)E * 32;
  dim3 scatGrid((unsigned)((scatThreads + B - 1) / B));
  dim3 nfGrid((unsigned)((NF + B - 1) / B));
  const int nTiles = (N + 15) / 16;
  dim3 gemmGrid((unsigned)((nTiles + 7) / 8));

  // 3) layer 1: aggregate x, normalize, GEMM -> h1
  k_zero<<<nfGrid, blk, 0, stream>>>(agg, NF);
  k_scatter<<<scatGrid, blk, 0, stream>>>(x, src, dst, agg, E);
  k_normalize<<<nfGrid, blk, 0, stream>>>(agg, cnt, NF);
  k_gemm_dual<<<gemmGrid, blk, 0, stream>>>(agg, WT1l, x, WT1r, b1, h1, nTiles);

  // 4) layer 2: aggregate h1, normalize, GEMM -> h2
  k_zero<<<nfGrid, blk, 0, stream>>>(agg, NF);
  k_scatter<<<scatGrid, blk, 0, stream>>>(h1, src, dst, agg, E);
  k_normalize<<<nfGrid, blk, 0, stream>>>(agg, cnt, NF);
  k_gemm_dual<<<gemmGrid, blk, 0, stream>>>(agg, WT2l, h1, WT2r, b2, h2, nTiles);

  // 5) JumpingKnowledge cat + final linear: out = relu([h1,h2] @ Wlin^T + blin)
  //    concat along K == two A sources; WlinT rows 0..127 pair with h1,
  //    rows 128..255 pair with h2.
  k_gemm_dual<<<gemmGrid, blk, 0, stream>>>(h1, WTlin, h2, WTlin + 128 * C128,
                                            blin, out, nTiles);
}
